// TFNPredictor_85753317032379
// MI455X (gfx1250) — matrix operand order repaired and checked by reading, not compiled, not run on hardware
//
#include <hip/hip_runtime.h>
#include <hip/hip_bf16.h>
#include <math.h>

typedef _Float16 v16h __attribute__((ext_vector_type(16)));
typedef _Float16 v8h  __attribute__((ext_vector_type(8)));
typedef float    v8f  __attribute__((ext_vector_type(8)));

union V16 { v16h v; v8h h[2]; };

// ============================================================
// Exact Wigner-3j-like tables (real spherical harmonic basis),
// computed once on device in double precision (matches the
// reference _cg / _q / _w3j pipeline).
// ============================================================
__device__ double d_fact(int n) { double r = 1.0; for (int i = 2; i <= n; ++i) r *= (double)i; return r; }

__device__ double d_cg(int j1, int m1, int j2, int m2, int j3, int m3) {
  if (m1 + m2 != m3) return 0.0;
  int kmin = 0;
  if (j2 - j3 - m1 > kmin) kmin = j2 - j3 - m1;
  if (j1 - j3 + m2 > kmin) kmin = j1 - j3 + m2;
  int kmax = j1 + j2 - j3;
  if (j1 - m1 < kmax) kmax = j1 - m1;
  if (j2 + m2 < kmax) kmax = j2 + m2;
  double s = 0.0;
  for (int k = kmin; k <= kmax; ++k) {
    double den = d_fact(k) * d_fact(j1 + j2 - j3 - k) * d_fact(j1 - m1 - k) *
                 d_fact(j2 + m2 - k) * d_fact(j3 - j2 + m1 + k) * d_fact(j3 - j1 - m2 + k);
    s += ((k & 1) ? -1.0 : 1.0) / den;
  }
  double pre = sqrt((double)(2 * j3 + 1) * d_fact(j1 + j2 - j3) * d_fact(j1 - j2 + j3) *
                    d_fact(-j1 + j2 + j3) / d_fact(j1 + j2 + j3 + 1));
  pre *= sqrt(d_fact(j1 + m1) * d_fact(j1 - m1) * d_fact(j2 + m2) * d_fact(j2 - m2) *
              d_fact(j3 + m3) * d_fact(j3 - m3));
  return pre * s;
}

struct DC { double re, im; };
__device__ DC d_q(int l, int r, int c) {
  int m = r - l;
  const double iv = 0.70710678118654752440;
  double re = 0.0, im = 0.0;
  if (m < 0) { int am = -m; if (c == l + am) re = iv; else if (c == l - am) im = -iv; }
  else if (m == 0) { if (c == l) re = 1.0; }
  else { double sg = (m & 1) ? -1.0 : 1.0; if (c == l + m) re = sg * iv; else if (c == l - m) im = sg * iv; }
  if (l == 1) { double t = re; re = im; im = -t; }      // *(-i)
  else if (l == 2) { re = -re; im = -im; }              // *(-1)
  DC o; o.re = re; o.im = im; return o;
}

__global__ void w3j_kernel(float* __restrict__ out) {
  int idx = threadIdx.x;
  if (idx >= 27) return;
  int l1 = idx / 9, l2 = (idx / 3) % 3, l3 = idx % 3;
  float* o = out + idx * 125;
  for (int t = 0; t < 125; ++t) o[t] = 0.0f;
  int lo = l1 - l2; if (lo < 0) lo = -lo;
  if (l3 < lo || l3 > l1 + l2) return;
  int d1 = 2 * l1 + 1, d2 = 2 * l2 + 1, d3 = 2 * l3 + 1;
  double Dr[125], Di[125];
  for (int i = 0; i < d1; ++i)
    for (int j = 0; j < d2; ++j)
      for (int k = 0; k < d3; ++k) {
        double sre = 0.0, sim = 0.0;
        for (int a = 0; a < d1; ++a)
          for (int b = 0; b < d2; ++b) {
            int m1 = a - l1, m2 = b - l2, m3 = m1 + m2;
            int c = m3 + l3;
            if (c < 0 || c >= d3) continue;
            double C = d_cg(l1, m1, l2, m2, l3, m3);
            if (C == 0.0) continue;
            DC qa = d_q(l1, a, i), qb = d_q(l2, b, j), qc = d_q(l3, c, k);
            qc.im = -qc.im;  // conj
            double tre = qa.re * qb.re - qa.im * qb.im;
            double tim = qa.re * qb.im + qa.im * qb.re;
            double ure = tre * qc.re - tim * qc.im;
            double uim = tre * qc.im + tim * qc.re;
            sre += C * ure; sim += C * uim;
          }
        Dr[(i * d2 + j) * d3 + k] = sre;
        Di[(i * d2 + j) * d3 + k] = sim;
      }
  int tot = d1 * d2 * d3;
  double ar = 0, ai = 0;
  for (int t = 0; t < tot; ++t) { ar += fabs(Dr[t]); ai += fabs(Di[t]); }
  const double* D = (ar >= ai) ? Dr : Di;
  double nrm = 0; for (int t = 0; t < tot; ++t) nrm += D[t] * D[t];
  nrm = sqrt(nrm);
  double inv = (nrm > 0.0) ? 1.0 / nrm : 0.0;
  for (int t = 0; t < tot; ++t) o[t] = (float)(D[t] * inv);
}

// ============================================================
// Tensor-product path metadata (hardcoded from the reference)
// ============================================================
__constant__ int TP_NP[3] = {3, 15, 3};
__constant__ int TP_L1[3][15] = {
  {0,0,0, 0,0,0,0,0,0,0,0,0,0,0,0},
  {0,0,0, 1,1,1,1,1,1, 2,2,2,2,2,2},
  {0,1,2, 0,0,0,0,0,0,0,0,0,0,0,0}};
__constant__ int TP_L2[3][15] = {
  {0,1,2, 0,0,0,0,0,0,0,0,0,0,0,0},
  {0,1,2, 0,1,1,1,2,2, 0,1,1,2,2,2},
  {0,1,2, 0,0,0,0,0,0,0,0,0,0,0,0}};
__constant__ int TP_L3[3][15] = {
  {0,1,2, 0,0,0,0,0,0,0,0,0,0,0,0},
  {0,1,2, 1,0,1,2,1,2, 2,1,2,0,1,2},
  {0,0,0, 0,0,0,0,0,0,0,0,0,0,0,0}};
__constant__ float TP_PW[3][15] = {
  {1.0f, 1.7320508f, 2.2360680f, 0,0,0,0,0,0,0,0,0,0,0,0},
  {0.57735027f,0.70710678f,0.91287093f,0.70710678f,0.57735027f,0.70710678f,0.91287093f,
   0.70710678f,0.91287093f,0.91287093f,0.70710678f,0.91287093f,0.57735027f,0.70710678f,0.91287093f},
  {0.57735027f,0.57735027f,0.57735027f, 0,0,0,0,0,0,0,0,0,0,0,0}};

// ============================================================
// Small utility kernels
// ============================================================
__global__ void zero_f32_kernel(float* __restrict__ p, long long n) {
  long long t = (long long)blockIdx.x * blockDim.x + threadIdx.x;
  if (t < n) p[t] = 0.0f;
}

__global__ void cvt_f16_kernel(const float* __restrict__ in, _Float16* __restrict__ out, int n) {
  int t = blockIdx.x * blockDim.x + threadIdx.x;
  if (t < n) out[t] = (_Float16)in[t];
}

__global__ void feat_init_kernel(const int* __restrict__ z, const float* __restrict__ emb,
                                 float* __restrict__ feat, int nN) {
  int t = blockIdx.x * blockDim.x + threadIdx.x;
  if (t >= nN * 64) return;
  int n = t >> 6, u = t & 63;
  feat[t] = emb[z[n] * 64 + u];
}

// ============================================================
// Per-edge geometry: spherical harmonics + radial embedding (f16)
// ============================================================
__global__ void edge_geom_kernel(const float* __restrict__ pos, const int* __restrict__ ei,
                                 int E, int e0, int ecount, int ecpad,
                                 float* __restrict__ shb, _Float16* __restrict__ emb) {
  int t = blockIdx.x * blockDim.x + threadIdx.x;
  if (t >= ecpad) return;
  if (t >= ecount) {           // zero-pad GEMM rows
    _Float16* em = emb + (size_t)t * 64;
    for (int i = 0; i < 64; ++i) em[i] = (_Float16)0.0f;
    return;
  }
  int e = e0 + t;
  int s = ei[e], d = ei[E + e];
  float dx = pos[3 * s + 0] - pos[3 * d + 0];
  float dy = pos[3 * s + 1] - pos[3 * d + 1];
  float dz = pos[3 * s + 2] - pos[3 * d + 2];
  float len = sqrtf(dx * dx + dy * dy + dz * dz) + 1e-8f;
  float x = dx / len, y = dy / len, z = dz / len;
  const float c0 = 0.28209479177387814f, c1 = 0.4886025119029199f;
  const float c2 = 0.6307831305050401f, s3 = 1.7320508075688772f;
  float* o = shb + (size_t)t * 9;
  o[0] = c0; o[1] = c1 * x; o[2] = c1 * y; o[3] = c1 * z;
  o[4] = c2 * s3 * x * z; o[5] = c2 * s3 * x * y;
  o[6] = c2 * (y * y - 0.5f * (x * x + z * z));
  o[7] = c2 * s3 * y * z; o[8] = c2 * (s3 * 0.5f) * (z * z - x * x);
  const float step = 5.0f / 63.0f;
  const float scale = 8.0f / 1.12f;   // RE^0.5 / 1.12
  _Float16* em = emb + (size_t)t * 64;
  for (int i = 0; i < 64; ++i) {
    float diff = (len - (float)i * step) / step;
    em[i] = (_Float16)(__expf(-diff * diff) * scale);
  }
}

// ============================================================
// WMMA GEMM: C[M x N] = act(A[M x K] @ B[K x N] + bias), N % 64 == 0
// - 256-thread block = 8 waves; block covers 8 M-tiles x one 64-col N panel.
// - B panel (K x 64 f16, <=16KB) staged in LDS once per block (ds_load path),
//   then each wave runs 4 register-blocked WMMAs per K-step (A fragment reused 4x).
// A fragment: lane = M row (mod 16), halves = K per ISA 16-bit A layout.
// B fragment: lane = N col (mod 16), halves = K in the same order as A.
// ============================================================
__global__ void gemm_wmma_kernel(const _Float16* __restrict__ A, const _Float16* __restrict__ B,
                                 const float* __restrict__ bias, void* __restrict__ Cout,
                                 int tilesM, int N, int K, int act, int storeHalf) {
  __shared__ _Float16 Bs[128 * 64];     // K <= 128
  int tid = threadIdx.x;
  int lane = tid & 31;
  int wave = tid >> 5;
  int nblk = blockIdx.y;                // 64-column panel index
  int ncol0 = nblk * 64;

  // --- cooperative stage of B[K x 64] panel into LDS (b128 copies) ---
  for (int idx = tid; idx < K * 8; idx += 256) {
    int row = idx >> 3, seg = idx & 7;
    *(v8h*)(Bs + row * 64 + seg * 8) = *(const v8h*)(B + (size_t)row * N + ncol0 + seg * 8);
  }
  __syncthreads();

  int tm = blockIdx.x * 8 + wave;
  if (tm >= tilesM) return;             // wave-uniform; no further barriers

  int rowA = tm * 16 + (lane & 15);
  int sel = (lane >> 4) << 3;           // lanes 16-31: K offset +8
  const _Float16* ap = A + (size_t)rowA * K;

  v8f acc0, acc1, acc2, acc3;
  for (int r = 0; r < 8; ++r) { acc0[r] = 0.f; acc1[r] = 0.f; acc2[r] = 0.f; acc3[r] = 0.f; }

  for (int k0 = 0; k0 < K; k0 += 32) {
    V16 a;
    a.h[0] = *(const v8h*)(ap + k0 + sel);          // K = k0+sel .. +7
    a.h[1] = *(const v8h*)(ap + k0 + 16 + sel);     // K = k0+16+sel .. +7
    const _Float16* bk = Bs + (k0 + sel) * 64 + (lane & 15);  // lane = N col, K per A layout
    V16 b0, b1, b2, b3;
    for (int h = 0; h < 8; ++h) {
      b0.h[0][h] = bk[h * 64 +  0]; b0.h[1][h] = bk[(16 + h) * 64 +  0];
      b1.h[0][h] = bk[h * 64 + 16]; b1.h[1][h] = bk[(16 + h) * 64 + 16];
      b2.h[0][h] = bk[h * 64 + 32]; b2.h[1][h] = bk[(16 + h) * 64 + 32];
      b3.h[0][h] = bk[h * 64 + 48]; b3.h[1][h] = bk[(16 + h) * 64 + 48];
    }
    acc0 = __builtin_amdgcn_wmma_f32_16x16x32_f16(false, a.v, false, b0.v, (short)0, acc0, false, false);
    acc1 = __builtin_amdgcn_wmma_f32_16x16x32_f16(false, a.v, false, b1.v, (short)0, acc1, false, false);
    acc2 = __builtin_amdgcn_wmma_f32_16x16x32_f16(false, a.v, false, b2.v, (short)0, acc2, false, false);
    acc3 = __builtin_amdgcn_wmma_f32_16x16x32_f16(false, a.v, false, b3.v, (short)0, acc3, false, false);
  }

  int mrow0 = tm * 16 + ((lane >> 4) << 3);         // D: M = r + 8*(lane>=16)
  int cbase = ncol0 + (lane & 15);
  v8f* accs[4] = {&acc0, &acc1, &acc2, &acc3};
  for (int s = 0; s < 4; ++s) {
    int col = cbase + s * 16;
    float bv = bias ? bias[col] : 0.0f;
    v8f& ac = *accs[s];
    for (int r = 0; r < 8; ++r) {
      float v = ac[r] + bv;
      if (act) v = v / (1.0f + __expf(-v));         // silu
      size_t idx = (size_t)(mrow0 + r) * N + col;
      if (storeHalf) ((_Float16*)Cout)[idx] = (_Float16)v;
      else           ((float*)Cout)[idx] = v;
    }
  }
}

// ============================================================
// Self-interaction: featOut[n, off+v*d+i] += (1/8) * sum_u featIn[n,off+u*d+i]*sc[u,v]
// WMMA over M=nodes, K=u=64, N=v=64, one launch-z per component i.
// ============================================================
__global__ void sc_wmma_kernel(const float* __restrict__ featIn, float* __restrict__ featOut,
                               const float* __restrict__ scw, int indim, int outdim,
                               int off_in, int off_out, int d) {
  int lane = threadIdx.x & 31;
  int tm = blockIdx.x, tn = blockIdx.y, comp = blockIdx.z;
  int m = tm * 16 + (lane & 15);
  int selA = (lane >> 4) << 3;
  const float* fr = featIn + (size_t)m * indim + off_in + comp;

  v8f acc; for (int r = 0; r < 8; ++r) acc[r] = 0.0f;
  for (int k0 = 0; k0 < 64; k0 += 32) {
    V16 a, b;
    const float* br = scw + (size_t)(k0 + selA) * 64 + tn * 16 + (lane & 15);  // lane = N col
#pragma unroll
    for (int h = 0; h < 8; ++h) {
      int ku = k0 + selA + h;
      int kv = k0 + 16 + selA + h;
      a.h[0][h] = (_Float16)fr[(size_t)ku * d];
      a.h[1][h] = (_Float16)fr[(size_t)kv * d];
      b.h[0][h] = (_Float16)br[h * 64];
      b.h[1][h] = (_Float16)br[(16 + h) * 64];
    }
    acc = __builtin_amdgcn_wmma_f32_16x16x32_f16(false, a.v, false, b.v,
                                                 (short)0, acc, false, false);
  }
  int col = tn * 16 + (lane & 15);
  int mrow0 = tm * 16 + ((lane >> 4) << 3);
  for (int r = 0; r < 8; ++r) {
    size_t idx = (size_t)(mrow0 + r) * outdim + off_out + (size_t)col * d + comp;
    featOut[idx] += acc[r] * 0.125f;                 // / R^0.5
  }
}

// ============================================================
// Tensor product + segment-sum scatter (one thread per edge,channel)
// ============================================================
__global__ void tp_scatter_kernel(const float* __restrict__ featIn, const float* __restrict__ shb,
                                  const float* __restrict__ wbuf, const int* __restrict__ ei,
                                  const float* __restrict__ w3j,
                                  int E, int e0, int ecount, int layer,
                                  int indim, int outdim, int wnum,
                                  float* __restrict__ featOut) {
  int t = blockIdx.x * blockDim.x + threadIdx.x;
  int te = t >> 6, u = t & 63;
  if (te >= ecount) return;
  int e = e0 + te;
  int src = ei[e], dst = ei[E + e];

  float shv[9];
  { const float* sp = shb + (size_t)te * 9;
#pragma unroll
    for (int j = 0; j < 9; ++j) shv[j] = sp[j]; }

  const float* fr = featIn + (size_t)src * indim;
  __builtin_prefetch((const void*)(featOut + (size_t)dst * outdim), 1, 3);
  float xin[9];
  xin[0] = fr[u];
  if (indim > 64) {
#pragma unroll
    for (int i = 0; i < 3; ++i) xin[1 + i] = fr[64 + u * 3 + i];
#pragma unroll
    for (int i = 0; i < 5; ++i) xin[4 + i] = fr[256 + u * 5 + i];
  } else {
#pragma unroll
    for (int i = 1; i < 9; ++i) xin[i] = 0.0f;
  }

  float acc[9];
#pragma unroll
  for (int i = 0; i < 9; ++i) acc[i] = 0.0f;

  const float* wrow = wbuf + (size_t)te * wnum;
  int np = TP_NP[layer];
  for (int k = 0; k < np; ++k) {
    int l1 = TP_L1[layer][k], l2 = TP_L2[layer][k], l3 = TP_L3[layer][k];
    float wk = wrow[k * 64 + u] * TP_PW[layer][k];
    int d1 = 2 * l1 + 1, d2 = 2 * l2 + 1, d3 = 2 * l3 + 1;
    const float* C = w3j + (((l1 * 3 + l2) * 3 + l3) * 125);
    int ib = l1 * l1, jb = l2 * l2, ob = l3 * l3;
    for (int i = 0; i < d1; ++i) {
      float xw = xin[ib + i] * wk;
      for (int j = 0; j < d2; ++j) {
        float xs = xw * shv[jb + j];
        const float* Cr = C + (i * d2 + j) * d3;
        for (int kk = 0; kk < d3; ++kk) acc[ob + kk] += xs * Cr[kk];
      }
    }
  }

  float* fo = featOut + (size_t)dst * outdim;
  atomicAdd(fo + u, acc[0]);
  if (outdim > 64) {
#pragma unroll
    for (int i = 0; i < 3; ++i) atomicAdd(fo + 64 + u * 3 + i, acc[1 + i]);
#pragma unroll
    for (int i = 0; i < 5; ++i) atomicAdd(fo + 256 + u * 5 + i, acc[4 + i]);
  }
}

// ============================================================
// Norm-gated activation (layers 0,1 output -> next layer input)
// ============================================================
__global__ void norm_act_kernel(const float* __restrict__ fin, float* __restrict__ fout, int nN) {
  int t = blockIdx.x * blockDim.x + threadIdx.x;
  if (t >= nN * 64) return;
  int n = t >> 6, u = t & 63;
  const float* r = fin + (size_t)n * 576;
  float* w = fout + (size_t)n * 576;
  float s = r[u];
  w[u] = s / (1.0f + __expf(-s));
  float v0 = r[64 + u * 3 + 0], v1 = r[64 + u * 3 + 1], v2 = r[64 + u * 3 + 2];
  float inner = (v0 * v0 + v1 * v1 + v2 * v2) * 0.57735027f;
  float g = 1.0f / (1.0f + __expf(-sqrtf(inner + 1e-8f)));
  w[64 + u * 3 + 0] = v0 * g; w[64 + u * 3 + 1] = v1 * g; w[64 + u * 3 + 2] = v2 * g;
  float q0 = r[256 + u * 5 + 0], q1 = r[256 + u * 5 + 1], q2 = r[256 + u * 5 + 2];
  float q3 = r[256 + u * 5 + 3], q4 = r[256 + u * 5 + 4];
  inner = (q0 * q0 + q1 * q1 + q2 * q2 + q3 * q3 + q4 * q4) * 0.4472136f;
  g = 1.0f / (1.0f + __expf(-sqrtf(inner + 1e-8f)));
  w[256 + u * 5 + 0] = q0 * g; w[256 + u * 5 + 1] = q1 * g; w[256 + u * 5 + 2] = q2 * g;
  w[256 + u * 5 + 3] = q3 * g; w[256 + u * 5 + 4] = q4 * g;
}

// ============================================================
// Pooling + output head
// ============================================================
__global__ void pool_kernel(const float* __restrict__ feat, const int* __restrict__ batch,
                            float* __restrict__ pooled, int nN) {
  int t = blockIdx.x * blockDim.x + threadIdx.x;
  if (t >= nN * 64) return;
  int n = t >> 6, u = t & 63;
  atomicAdd(&pooled[batch[n] * 64 + u], feat[t]);
}

__global__ void head_kernel(const float* __restrict__ pooled,
                            const float* __restrict__ w1, const float* __restrict__ b1,
                            const float* __restrict__ w2, const float* __restrict__ b2,
                            const float* __restrict__ w3, const float* __restrict__ b3,
                            float* __restrict__ out) {
  __shared__ float h1[16 * 128];
  __shared__ float h2[16 * 64];
  int t = threadIdx.x;
  for (int idx = t; idx < 16 * 128; idx += 256) {
    int g = idx >> 7, j = idx & 127;
    float s = b1[j];
    for (int k = 0; k < 64; ++k) s += pooled[g * 64 + k] * w1[k * 128 + j];
    h1[idx] = s / (1.0f + __expf(-s));
  }
  __syncthreads();
  for (int idx = t; idx < 16 * 64; idx += 256) {
    int g = idx >> 6, j = idx & 63;
    float s = b2[j];
    for (int k = 0; k < 128; ++k) s += h1[g * 128 + k] * w2[k * 64 + j];
    h2[idx] = s / (1.0f + __expf(-s));
  }
  __syncthreads();
  if (t < 16) {
    float s = b3[0];
    for (int k = 0; k < 64; ++k) s += h2[t * 64 + k] * w3[k];
    out[t] = 1.0f / (1.0f + __expf(-s));
  }
}

// ============================================================
// Host orchestration
// ============================================================
static inline int cdivi(long long a, long long b) { return (int)((a + b - 1) / b); }

extern "C" void kernel_launch(void* const* d_in, const int* in_sizes, int n_in,
                              void* d_out, int out_size, void* d_ws, size_t ws_size,
                              hipStream_t stream) {
  // d_in flatten order (jax pytree, dict keys sorted):
  // 0:z 1:pos 2:batch 3:edge_index 4:emb
  // layer0: 5 fc1_b 6 fc1_w 7 fc2_b 8 fc2_w 9 fc3_b 10 fc3_w 11 sc'0'
  // layer1: 12..17 same, 18 sc'0' 19 sc'1' 20 sc'2'
  // layer2: 21..26 same, 27 sc'0'
  // out: 28 b1 29 b2 30 b3 31 w1 32 w2 33 w3
  const int*   z    = (const int*)d_in[0];
  const float* pos  = (const float*)d_in[1];
  const int*   bat  = (const int*)d_in[2];
  const int*   ei   = (const int*)d_in[3];
  const float* embp = (const float*)d_in[4];
  int Nn = in_sizes[0];
  int E  = in_sizes[3] / 2;

  const float* fc1b[3] = {(const float*)d_in[5],  (const float*)d_in[12], (const float*)d_in[21]};
  const float* fc1w[3] = {(const float*)d_in[6],  (const float*)d_in[13], (const float*)d_in[22]};
  const float* fc2b[3] = {(const float*)d_in[7],  (const float*)d_in[14], (const float*)d_in[23]};
  const float* fc2w[3] = {(const float*)d_in[8],  (const float*)d_in[15], (const float*)d_in[24]};
  const float* fc3b[3] = {(const float*)d_in[9],  (const float*)d_in[16], (const float*)d_in[25]};
  const float* fc3w[3] = {(const float*)d_in[10], (const float*)d_in[17], (const float*)d_in[26]};
  const float* sc0[3]  = {(const float*)d_in[11], (const float*)d_in[18], (const float*)d_in[27]};
  const float* sc1 = (const float*)d_in[19];
  const float* sc2 = (const float*)d_in[20];
  const float* hb1 = (const float*)d_in[28];
  const float* hb2 = (const float*)d_in[29];
  const float* hb3 = (const float*)d_in[30];
  const float* hw1 = (const float*)d_in[31];
  const float* hw2 = (const float*)d_in[32];
  const float* hw3 = (const float*)d_in[33];

  // workspace carve
  char* wp = (char*)d_ws;
  auto carve = [&](size_t bytes) -> void* {
    void* r = (void*)wp; wp += (bytes + 255) & ~(size_t)255; return r;
  };
  float*    w3jbuf = (float*)carve(27 * 125 * sizeof(float));
  float*    featA  = (float*)carve((size_t)Nn * 576 * 4);
  float*    featB  = (float*)carve((size_t)Nn * 576 * 4);
  float*    pooled = (float*)carve(16 * 64 * 4);
  _Float16* wt1    = (_Float16*)carve(64 * 128 * 2);
  _Float16* wt2    = (_Float16*)carve(128 * 64 * 2);
  _Float16* wt3    = (_Float16*)carve(64 * 960 * 2);

  size_t used = (size_t)(wp - (char*)d_ws);
  const size_t perEdge = 9 * 4 + 64 * 2 + 128 * 2 + 64 * 2 + 960 * 4; // 4388 B/edge chunk
  int EC = 8192;
  if (ws_size > used + 8192) {
    long long cap = ((long long)ws_size - (long long)used - 8192) / (long long)perEdge;
    if (cap < EC) EC = (int)cap;
  } else EC = 16;
  EC &= ~15; if (EC < 16) EC = 16;

  float*    shbuf = (float*)carve((size_t)EC * 9 * 4);
  _Float16* embuf = (_Float16*)carve((size_t)EC * 64 * 2);
  _Float16* h1b   = (_Float16*)carve((size_t)EC * 128 * 2);
  _Float16* h2b   = (_Float16*)carve((size_t)EC * 64 * 2);
  float*    wbuf  = (float*)carve((size_t)EC * 960 * 4);

  const int WNUMS[3]  = {192, 960, 192};
  const int INDIM[3]  = {64, 576, 576};
  const int OUTDIM[3] = {576, 576, 64};

  w3j_kernel<<<1, 32, 0, stream>>>(w3jbuf);
  feat_init_kernel<<<cdivi((long long)Nn * 64, 256), 256, 0, stream>>>(z, embp, featA, Nn);

  for (int l = 0; l < 3; ++l) {
    cvt_f16_kernel<<<cdivi(64 * 128, 256), 256, 0, stream>>>(fc1w[l], wt1, 64 * 128);
    cvt_f16_kernel<<<cdivi(128 * 64, 256), 256, 0, stream>>>(fc2w[l], wt2, 128 * 64);
    cvt_f16_kernel<<<cdivi(64 * WNUMS[l], 256), 256, 0, stream>>>(fc3w[l], wt3, 64 * WNUMS[l]);
    zero_f32_kernel<<<cdivi((long long)Nn * OUTDIM[l], 256), 256, 0, stream>>>(
        featB, (long long)Nn * OUTDIM[l]);

    for (int e0 = 0; e0 < E; e0 += EC) {
      int ec = (E - e0 < EC) ? (E - e0) : EC;
      int ecp = (ec + 15) & ~15;
      edge_geom_kernel<<<cdivi(ecp, 256), 256, 0, stream>>>(pos, ei, E, e0, ec, ecp, shbuf, embuf);
      int tm = ecp >> 4;
      // grid: x = M-tile groups of 8 waves, y = 64-column N panels
      gemm_wmma_kernel<<<dim3(cdivi(tm, 8), 128 / 64), 256, 0, stream>>>(
          embuf, wt1, fc1b[l], h1b, tm, 128, 64, 1, 1);
      gemm_wmma_kernel<<<dim3(cdivi(tm, 8), 64 / 64), 256, 0, stream>>>(
          h1b, wt2, fc2b[l], h2b, tm, 64, 128, 1, 1);
      gemm_wmma_kernel<<<dim3(cdivi(tm, 8), WNUMS[l] / 64), 256, 0, stream>>>(
          h2b, wt3, fc3b[l], wbuf, tm, WNUMS[l], 64, 0, 0);
      tp_scatter_kernel<<<cdivi((long long)ec * 64, 256), 256, 0, stream>>>(
          featA, shbuf, wbuf, ei, w3jbuf, E, e0, ec, l, INDIM[l], OUTDIM[l], WNUMS[l], featB);
    }

    int tmn = Nn >> 4;  // node tiles (10000/16 = 625)
    if (l == 0) {
      sc_wmma_kernel<<<dim3(tmn, 4, 1), 32, 0, stream>>>(featA, featB, sc0[0], 64, 576, 0, 0, 1);
    } else if (l == 1) {
      sc_wmma_kernel<<<dim3(tmn, 4, 1), 32, 0, stream>>>(featA, featB, sc0[1], 576, 576, 0, 0, 1);
      sc_wmma_kernel<<<dim3(tmn, 4, 3), 32, 0, stream>>>(featA, featB, sc1, 576, 576, 64, 64, 3);
      sc_wmma_kernel<<<dim3(tmn, 4, 5), 32, 0, stream>>>(featA, featB, sc2, 576, 576, 256, 256, 5);
    } else {
      sc_wmma_kernel<<<dim3(tmn, 4, 1), 32, 0, stream>>>(featA, featB, sc0[2], 576, 64, 0, 0, 1);
    }

    if (l < 2)
      norm_act_kernel<<<cdivi((long long)Nn * 64, 256), 256, 0, stream>>>(featB, featA, Nn);
  }

  zero_f32_kernel<<<4, 256, 0, stream>>>(pooled, 1024);
  pool_kernel<<<cdivi((long long)Nn * 64, 256), 256, 0, stream>>>(featB, bat, pooled, Nn);
  head_kernel<<<1, 256, 0, stream>>>(pooled, hw1, hb1, hw2, hb2, hw3, hb3, (float*)d_out);

  (void)n_in; (void)out_size;
}
